// GNModule_31233002176630
// MI455X (gfx1250) — matrix operand order repaired and checked by reading, not compiled, not run on hardware
//
#include <hip/hip_runtime.h>
#include <hip/hip_bf16.h>

#define N_NODES 100000
#define N_EDGES 1600000
#define DIM     128
#define EPS_BN  1e-5f

typedef __attribute__((ext_vector_type(16))) __bf16 v16bf;
typedef __attribute__((ext_vector_type(8)))  float  v8f;

// ---------------- degree: deg[row[e]] += 1 ----------------
__global__ void k_degree(const int* __restrict__ row, float* __restrict__ deg) {
    int e = blockIdx.x * blockDim.x + threadIdx.x;
    if (e < N_EDGES) atomicAdd(&deg[row[e]], 1.0f);
}

// ---------------- dinv = deg>0 ? rsqrt(deg) : 0 (in place) ----------------
__global__ void k_dinv(float* __restrict__ deg) {
    int i = blockIdx.x * blockDim.x + threadIdx.x;
    if (i < N_NODES) {
        float d = deg[i];
        deg[i] = (d > 0.0f) ? rsqrtf(d) : 0.0f;
    }
}

// ------------- propagation: dst[row] += ew * src[col], one wave32 per edge -------------
__global__ void __launch_bounds__(256)
k_prop(const int* __restrict__ row, const int* __restrict__ col,
       const float* __restrict__ dinv, const float* __restrict__ src,
       float* __restrict__ dst) {
    int wave = (int)((blockIdx.x * (unsigned)blockDim.x + threadIdx.x) >> 5);
    int lane = threadIdx.x & 31;
    if (wave >= N_EDGES) return;
    int r = row[wave], c = col[wave];
    float ew = -dinv[r] * dinv[c];                    // uniform across wave
    const float4* s4 = (const float4*)(src + (size_t)c * DIM);
    float4 v = s4[lane];                              // 512B coalesced gather per edge
    float* d = dst + (size_t)r * DIM + lane * 4;
    atomicAdd(d + 0, ew * v.x);
    atomicAdd(d + 1, ew * v.y);
    atomicAdd(d + 2, ew * v.z);
    atomicAdd(d + 3, ew * v.w);
}

// ------------- W pre-pack: fragment-major bf16 hi/lo, W0-W2 folded -------------
// Record layout: idx t = ((m*4+kc)*8+wv)*32+lane -> 64B {v16bf hi, v16bf lo}.
// B fragment convention (32x16, K x N): lane%16 = N, slot e<8 -> K = 8*(lane/16)+e,
// slot e>=8 -> K = 16 + 8*(lane/16) + (e-8).
__global__ void k_packW(const float* __restrict__ W, v16bf* __restrict__ pack) {
    int t    = blockIdx.x * blockDim.x + threadIdx.x;   // 0..3071
    int lane = t & 31;
    int wv   = (t >> 5) & 7;
    int kc   = (t >> 8) & 3;
    int m    = t >> 10;
    int hi   = lane >> 4;
    int o    = wv * 16 + (lane & 15);
    v16bf h, l;
    #pragma unroll
    for (int e = 0; e < 16; ++e) {
        int k = kc * 32 + ((e < 8) ? e : 8 + e) + 8 * hi;
        float f = W[m * (DIM * DIM) + k * DIM + o];
        if (m == 0) f -= W[2 * (DIM * DIM) + k * DIM + o];  // W0' = W0 - W2
        __bf16 hb = (__bf16)f;
        h[e] = hb;
        l[e] = (__bf16)(f - (float)hb);
    }
    pack[t * 2]     = h;
    pack[t * 2 + 1] = l;
}

// ------------- fused Chebyshev GEMM (bf16-split WMMA) + bias + PReLU + BN stats -------------
// out = tx0@(W0-W2) + tx1@W1 + (2*p2)@W2 + bias ; tx2 = 2*p2 - tx0 folded algebraically.
__global__ void __launch_bounds__(256)
k_cheb_wmma(const float* __restrict__ x, const float* __restrict__ tx1,
            const float* __restrict__ p2, const v16bf* __restrict__ wpack,
            const float* __restrict__ bias, const float* __restrict__ prelu_a,
            float* __restrict__ out, float* __restrict__ sumArr,
            float* __restrict__ sqArr) {
    const int lane = threadIdx.x & 31;
    const int wv   = threadIdx.x >> 5;     // 0..7 : 16-wide output-column tile
    const int lm   = lane & 15;
    const int hi   = lane >> 4;
    const int row0 = blockIdx.x * 16;      // N_NODES % 16 == 0
    const int node = row0 + lm;            // A-matrix row owned by this lane
    const int o    = wv * 16 + lm;         // C/D column owned by this lane

    // A fragments are identical across the 8 output tiles -> build once per
    // block, pre-split into bf16 hi/lo, staged in LDS (24KB of 320KB/WGP).
    __shared__ v16bf ldsA[12 * 32 * 2];    // [combo c=m*4+kc][lane]{hi,lo}

    for (int c = wv; c < 12; c += 8) {     // 12 combos over 8 waves
        const int m  = c >> 2;
        const int kc = c & 3;
        const float* aSrc = (m == 0) ? x : (m == 1) ? tx1 : p2;
        const float  scl  = (m == 2) ? 2.0f : 1.0f;          // 2*p2 fold
        const float* aRow = aSrc + (size_t)node * DIM + kc * 32;
        v16bf aHi, aLo;
        // A fragment: 16x32 bf16 layout (ISA 7.12.2): per-lane K pairs,
        // +8 K-offset for high lane half, +16 for upper 4 VGPRs.
        #pragma unroll
        for (int v = 0; v < 8; ++v) {
            int kb = ((v < 4) ? 2 * v : 16 + 2 * (v - 4)) + 8 * hi;
            float f0 = aRow[kb]     * scl;
            float f1 = aRow[kb + 1] * scl;
            __bf16 h0 = (__bf16)f0, h1 = (__bf16)f1;
            aHi[2 * v]     = h0; aLo[2 * v]     = (__bf16)(f0 - (float)h0);
            aHi[2 * v + 1] = h1; aLo[2 * v + 1] = (__bf16)(f1 - (float)h1);
        }
        ldsA[(c * 32 + lane) * 2]     = aHi;
        ldsA[(c * 32 + lane) * 2 + 1] = aLo;
    }
    __syncthreads();

    v8f acc = {0.f, 0.f, 0.f, 0.f, 0.f, 0.f, 0.f, 0.f};
    const v16bf* bp = wpack + ((size_t)wv * 32 + lane) * 2;  // this wave's B records

    for (int c = 0; c < 12; ++c) {         // K = 384 total, 32 per step
        v16bf aHi = ldsA[(c * 32 + lane) * 2];               // ds_load_b128 x2
        v16bf aLo = ldsA[(c * 32 + lane) * 2 + 1];
        v16bf bHi = bp[c * 512];                             // contiguous 64B record
        v16bf bLo = bp[c * 512 + 1];
        // 2-term bf16 split: hi*hi + hi*lo + lo*hi  (~f32 accuracy)
        acc = __builtin_amdgcn_wmma_f32_16x16x32_bf16(false, aHi, false, bHi, (short)0, acc, false, false);
        acc = __builtin_amdgcn_wmma_f32_16x16x32_bf16(false, aHi, false, bLo, (short)0, acc, false, false);
        acc = __builtin_amdgcn_wmma_f32_16x16x32_bf16(false, aLo, false, bHi, (short)0, acc, false, false);
    }

    // Epilogue: bias + PReLU, store pre-BN output, accumulate per-feature stats.
    const float bv = bias[o];
    const float pa = prelu_a[0];
    float s = 0.f, sq = 0.f;
    #pragma unroll
    for (int r = 0; r < 8; ++r) {
        float v = acc[r] + bv;                  // C/D row = row0 + r + 8*hi
        v = (v > 0.f) ? v : pa * v;
        out[(size_t)(row0 + r + 8 * hi) * DIM + o] = v;
        s += v; sq += v * v;
    }
    s  += __shfl_xor(s, 16);                    // combine the two row-halves (same o)
    sq += __shfl_xor(sq, 16);
    if (hi == 0) {
        atomicAdd(&sumArr[o], s);
        atomicAdd(&sqArr[o],  sq);
    }
}

// ---------------- BatchNorm finalize (biased variance) ----------------
__global__ void k_bn(float* __restrict__ out, const float* __restrict__ sumArr,
                     const float* __restrict__ sqArr, const float* __restrict__ gamma,
                     const float* __restrict__ beta) {
    size_t idx = (size_t)blockIdx.x * blockDim.x + threadIdx.x;
    if (idx >= (size_t)N_NODES * DIM) return;
    int f = (int)(idx & (DIM - 1));
    const float invN = 1.0f / (float)N_NODES;
    float mean = sumArr[f] * invN;
    float var  = sqArr[f] * invN - mean * mean;
    float inv  = rsqrtf(var + EPS_BN);
    out[idx] = gamma[f] * (out[idx] - mean) * inv + beta[f];
}

extern "C" void kernel_launch(void* const* d_in, const int* in_sizes, int n_in,
                              void* d_out, int out_size, void* d_ws, size_t ws_size,
                              hipStream_t stream) {
    (void)in_sizes; (void)n_in; (void)out_size; (void)ws_size;
    const float* x     = (const float*)d_in[0];
    const int*   ei    = (const int*)d_in[1];
    const float* W     = (const float*)d_in[2];
    const float* bias  = (const float*)d_in[3];
    const float* pa    = (const float*)d_in[4];
    const float* gamma = (const float*)d_in[5];
    const float* beta  = (const float*)d_in[6];
    float* out = (float*)d_out;
    const int* row = ei;                // edge_index[0]
    const int* col = ei + N_EDGES;      // edge_index[1]

    // Workspace: [deg/dinv][sum][sumsq][tx1][p2] (zeroed) + [wpack] (fully overwritten)
    char* ws = (char*)d_ws;
    size_t off = 0;
    float* deg    = (float*)(ws + off); off += ((size_t)N_NODES * 4 + 255) & ~(size_t)255;
    float* sumArr = (float*)(ws + off); off += 512;
    float* sqArr  = (float*)(ws + off); off += 512;
    float* tx1    = (float*)(ws + off); off += (size_t)N_NODES * DIM * 4;
    float* p2     = (float*)(ws + off); off += (size_t)N_NODES * DIM * 4;
    const size_t zeroBytes = off;
    v16bf* wpack  = (v16bf*)(ws + off); off += (size_t)3072 * 64;   // 196KB, L2-resident

    hipMemsetAsync(d_ws, 0, zeroBytes, stream);  // capture-safe memset node

    k_degree<<<(N_EDGES + 255) / 256, 256, 0, stream>>>(row, deg);
    k_dinv  <<<(N_NODES + 255) / 256, 256, 0, stream>>>(deg);
    k_packW <<<12, 256, 0, stream>>>(W, wpack);
    k_prop  <<<N_EDGES / 8, 256, 0, stream>>>(row, col, deg, x,   tx1);  // tx1 = L_hat @ x
    k_prop  <<<N_EDGES / 8, 256, 0, stream>>>(row, col, deg, tx1, p2);   // p2  = L_hat @ tx1
    k_cheb_wmma<<<N_NODES / 16, 256, 0, stream>>>(x, tx1, p2, wpack, bias, pa,
                                                  out, sumArr, sqArr);
    k_bn<<<(int)(((size_t)N_NODES * DIM + 255) / 256), 256, 0, stream>>>(out, sumArr, sqArr,
                                                                         gamma, beta);
}
